// W8A8DynamicLinearStatic_78331613545235
// MI455X (gfx1250) — 8-bit matrix operand order repaired and checked by reading, compiled, not run on hardware
//
#include <hip/hip_runtime.h>
#include <hip/hip_bf16.h>

typedef __attribute__((ext_vector_type(8))) int v8i;

#define MDIM 4096
#define KDIM 4096
#define NDIM 11008

#if defined(__has_builtin)
#if __has_builtin(__builtin_amdgcn_global_load_async_to_lds_b128)
#define USE_ASYNC_LDS 1
#endif
#endif
#ifndef USE_ASYNC_LDS
#define USE_ASYNC_LDS 0
#endif

#if USE_ASYNC_LDS
typedef int v4i_b __attribute__((vector_size(16)));
typedef __attribute__((address_space(1))) v4i_b* as1_v4i;
typedef __attribute__((address_space(3))) v4i_b* as3_v4i;

__device__ __forceinline__ void async_cp16(const void* g, void* l) {
  __builtin_amdgcn_global_load_async_to_lds_b128(
      (as1_v4i)(v4i_b*)(void*)(const_cast<void*>(g)), (as3_v4i)(v4i_b*)l, 0, 0);
}
__device__ __forceinline__ void async_wait0() {
#if __has_builtin(__builtin_amdgcn_s_wait_asynccnt)
  __builtin_amdgcn_s_wait_asynccnt(0);
#else
  asm volatile("s_wait_asynccnt 0x0" ::: "memory");
#endif
}
#endif

__global__ __launch_bounds__(256) void quant_rows_kernel(
    const float* __restrict__ x,
    signed char* __restrict__ xq,
    float* __restrict__ xscale) {
  const int row = blockIdx.x;
  const int t = threadIdx.x;
  const float4* xr4 = (const float4*)(x + (size_t)row * KDIM);

  float4 v[4];
  float mx = 0.0f;
#pragma unroll
  for (int i = 0; i < 4; ++i) {
    v[i] = xr4[t * 4 + i];
    mx = fmaxf(mx, fmaxf(fmaxf(fabsf(v[i].x), fabsf(v[i].y)),
                         fmaxf(fabsf(v[i].z), fabsf(v[i].w))));
  }

  __shared__ float red[256];
  red[t] = mx;
  __syncthreads();
#pragma unroll
  for (int s = 128; s > 0; s >>= 1) {
    if (t < s) red[t] = fmaxf(red[t], red[t + s]);
    __syncthreads();
  }
  const float scale = fmaxf(red[0] * (1.0f / 127.0f), 1e-8f);
  if (t == 0) xscale[row] = scale;
  const float inv = 1.0f / scale;

  int packed[4];
#pragma unroll
  for (int i = 0; i < 4; ++i) {
    int b0 = (int)rintf(fminf(fmaxf(v[i].x * inv, -128.0f), 127.0f));
    int b1 = (int)rintf(fminf(fmaxf(v[i].y * inv, -128.0f), 127.0f));
    int b2 = (int)rintf(fminf(fmaxf(v[i].z * inv, -128.0f), 127.0f));
    int b3 = (int)rintf(fminf(fmaxf(v[i].w * inv, -128.0f), 127.0f));
    packed[i] = (b0 & 255) | ((b1 & 255) << 8) | ((b2 & 255) << 16) |
                ((b3 & 255) << 24);
  }
  int4 r;
  r.x = packed[0]; r.y = packed[1]; r.z = packed[2]; r.w = packed[3];
  ((int4*)(xq + (size_t)row * KDIM))[t] = r;
}

__global__ __launch_bounds__(256) void w8a8_wmma_gemm_kernel(
    const signed char* __restrict__ Aq,
    const signed char* __restrict__ Wq,
    const float* __restrict__ xscale,
    const float* __restrict__ wscale,
    const float* __restrict__ bias,
    float* __restrict__ out) {
  constexpr int BM = 128, BN = 128, BK = 128;
  constexpr int LSTR = 144;

  __shared__ signed char sA[2][BM * LSTR];
  __shared__ signed char sB[2][BN * LSTR];

  const int tid = threadIdx.x;
  const int lane = tid & 31;
  const int wave = tid >> 5;
  const int wm = wave >> 1;
  const int wn = wave & 1;
  const int m0 = blockIdx.y * BM;
  const int n0 = blockIdx.x * BN;

  const int srow = tid >> 3;
  const int soff = (tid & 7) * 16;
  const signed char* Ag[4];
  const signed char* Bg[4];
  int ldso[4];
#pragma unroll
  for (int i = 0; i < 4; ++i) {
    const int r = srow + 32 * i;
    Ag[i] = Aq + (size_t)(m0 + r) * KDIM + soff;
    Bg[i] = Wq + (size_t)(n0 + r) * KDIM + soff;
    ldso[i] = r * LSTR + soff;
  }

  const int aoff = (lane >= 16) ? 8 : 0;
  const int boff = (lane >= 16) ? 8 : 0;
  const int mrow_base = wm * 32 + (lane & 15);
  const int nrow_base = wn * 64 + (lane & 15);

  v8i acc[2][4];
#pragma unroll
  for (int i = 0; i < 2; ++i)
#pragma unroll
    for (int j = 0; j < 4; ++j) {
      v8i z = {0, 0, 0, 0, 0, 0, 0, 0};
      acc[i][j] = z;
    }

  constexpr int NK = KDIM / BK;

#if USE_ASYNC_LDS
#pragma unroll
  for (int i = 0; i < 4; ++i) {
    async_cp16(Ag[i], &sA[0][ldso[i]]);
    async_cp16(Bg[i], &sB[0][ldso[i]]);
  }
  async_wait0();
  __syncthreads();
#else
  int4 pre[8];
#pragma unroll
  for (int i = 0; i < 4; ++i) {
    pre[i] = *(const int4*)Ag[i];
    pre[4 + i] = *(const int4*)Bg[i];
  }
#pragma unroll
  for (int i = 0; i < 4; ++i) {
    *(int4*)&sA[0][ldso[i]] = pre[i];
    *(int4*)&sB[0][ldso[i]] = pre[4 + i];
  }
  __syncthreads();
#endif

  for (int kt = 0; kt < NK; ++kt) {
    const int cur = kt & 1;
    const int nxt = cur ^ 1;
    const size_t koff = (size_t)(kt + 1) * BK;

#if USE_ASYNC_LDS
    if (kt + 1 < NK) {
#pragma unroll
      for (int i = 0; i < 4; ++i) {
        async_cp16(Ag[i] + koff, &sA[nxt][ldso[i]]);
        async_cp16(Bg[i] + koff, &sB[nxt][ldso[i]]);
      }
    }
#else
    if (kt + 1 < NK) {
#pragma unroll
      for (int i = 0; i < 4; ++i) {
        pre[i] = *(const int4*)(Ag[i] + koff);
        pre[4 + i] = *(const int4*)(Bg[i] + koff);
      }
    }
#endif

#pragma unroll
    for (int s = 0; s < 2; ++s) {
      const int sb = s * 64;
      v8i afrag[2], bfrag[4];
#pragma unroll
      for (int mt = 0; mt < 2; ++mt) {
        const signed char* p =
            &sA[cur][(mrow_base + mt * 16) * LSTR + sb + aoff];
        int2 q0 = *(const int2*)(p + 0);
        int2 q1 = *(const int2*)(p + 16);
        int2 q2 = *(const int2*)(p + 32);
        int2 q3 = *(const int2*)(p + 48);
        v8i a = {q0.x, q0.y, q1.x, q1.y, q2.x, q2.y, q3.x, q3.y};
        afrag[mt] = a;
      }
#pragma unroll
      for (int nt = 0; nt < 4; ++nt) {
        const signed char* p =
            &sB[cur][(nrow_base + nt * 16) * LSTR + sb + boff];
        int2 q0 = *(const int2*)(p + 0);
        int2 q1 = *(const int2*)(p + 16);
        int2 q2 = *(const int2*)(p + 32);
        int2 q3 = *(const int2*)(p + 48);
        v8i b = {q0.x, q0.y, q1.x, q1.y, q2.x, q2.y, q3.x, q3.y};
        bfrag[nt] = b;
      }
#pragma unroll
      for (int mt = 0; mt < 2; ++mt)
#pragma unroll
        for (int nt = 0; nt < 4; ++nt)
          acc[mt][nt] = __builtin_amdgcn_wmma_i32_16x16x64_iu8(
               true, afrag[mt],  true, bfrag[nt],
              acc[mt][nt],  false,  false);
    }

#if USE_ASYNC_LDS
    async_wait0();
    __syncthreads();
#else
    __syncthreads();
    if (kt + 1 < NK) {
#pragma unroll
      for (int i = 0; i < 4; ++i) {
        *(int4*)&sA[nxt][ldso[i]] = pre[i];
        *(int4*)&sB[nxt][ldso[i]] = pre[4 + i];
      }
      __syncthreads();
    }
#endif
  }

  const int rofs = (lane >= 16) ? 8 : 0;
  const int coll = lane & 15;
#pragma unroll
  for (int mt = 0; mt < 2; ++mt) {
    float xs[8];
#pragma unroll
    for (int r = 0; r < 8; ++r)
      xs[r] = xscale[m0 + wm * 32 + mt * 16 + rofs + r];
#pragma unroll
    for (int nt = 0; nt < 4; ++nt) {
      const int gcol = n0 + wn * 64 + nt * 16 + coll;
      const float ws = wscale[gcol];
      const float bs = bias[gcol];
#pragma unroll
      for (int r = 0; r < 8; ++r) {
        const int grow = m0 + wm * 32 + mt * 16 + rofs + r;
        out[(size_t)grow * NDIM + gcol] =
            (float)acc[mt][nt][r] * xs[r] * ws + bs;
      }
    }
  }
}

extern "C" void kernel_launch(void* const* d_in, const int* in_sizes, int n_in,
                              void* d_out, int out_size, void* d_ws,
                              size_t ws_size, hipStream_t stream) {
  (void)in_sizes; (void)n_in; (void)out_size; (void)ws_size;
  const float* x = (const float*)d_in[0];
  const signed char* w = (const signed char*)d_in[1];
  const float* wsc = (const float*)d_in[2];
  const float* bias = (const float*)d_in[3];
  float* out = (float*)d_out;

  signed char* xq = (signed char*)d_ws;
  float* xs = (float*)((char*)d_ws + (size_t)MDIM * KDIM);

  quant_rows_kernel<<<MDIM, 256, 0, stream>>>(x, xq, xs);

  dim3 grid(NDIM / 128, MDIM / 128);
  w8a8_wmma_gemm_kernel<<<grid, 256, 0, stream>>>(xq, w, xs, wsc, bias, out);
}
